// GraphTransformer_27986006901508
// MI455X (gfx1250) — compile-verified, matrix operand order checked by reading, not run on hardware
//
#include <hip/hip_runtime.h>
#include <hip/hip_bf16.h>
#include <math.h>

// ---------------------------------------------------------------------------
// Types for WMMA operands (CDNA5 / gfx1250, wave32)
// ---------------------------------------------------------------------------
typedef __attribute__((ext_vector_type(2)))  float   v2f;
typedef __attribute__((ext_vector_type(8)))  float   v8f;
typedef __attribute__((ext_vector_type(8)))  __bf16  v8bf;
typedef __attribute__((ext_vector_type(16))) __bf16  v16bf;

// Exact parameter types for __builtin_amdgcn_global_load_async_to_lds_b128:
// (AS1 int4*, AS3 int4*, imm offset, imm cpol) — int4 is GCC vector_size(16).
typedef int gt_v4i __attribute__((vector_size(16)));
typedef __attribute__((address_space(1))) gt_v4i* gt_gp;
typedef __attribute__((address_space(3))) gt_v4i* gt_lp;

#define NU     8192
#define NI     8192
#define DMODEL 32
#define KEYCH  32
#define HEADS  4
#define HK     8      // per-head key channels
#define BIASC  0.1f

#define UTILE  32                 // users per main-loop chunk
#define ITILE  32                 // items per wave (two 16-wide N subtiles)
#define ZSPLIT 2                  // u-range split for parallelism
#define NCHUNK ((NU / ZSPLIT) / UTILE)

// ---------------------------------------------------------------------------
// Phase 1: dense layer  out[n,c] = E[n,:] @ W[:,c] + b[c]   (n x 32, K=32)
// ---------------------------------------------------------------------------
__global__ void gt_dense_kernel(const float* __restrict__ E,
                                const float* __restrict__ W,
                                const float* __restrict__ b,
                                float* __restrict__ out) {
    int t = blockIdx.x * blockDim.x + threadIdx.x;   // N*32 threads
    int n = t >> 5, c = t & 31;
    const float* er = E + n * DMODEL;
    float acc = b[c];
#pragma unroll
    for (int j = 0; j < DMODEL; ++j)
        acc = fmaf(er[j], W[j * KEYCH + c], acc);
    out[t] = acc;
}

// ---------------------------------------------------------------------------
// Phase 2: column softmax over users, in place on k[NU,32].
// ---------------------------------------------------------------------------
__global__ void gt_col_softmax_kernel(float* __restrict__ k) {
    __shared__ float red[1024];
    const int c = threadIdx.x & 31;
    const int g = threadIdx.x >> 5;
    float m = -INFINITY;
    for (int r = g; r < NU; r += 32) m = fmaxf(m, k[r * KEYCH + c]);
    red[threadIdx.x] = m;
    __syncthreads();
    for (int s = 16; s > 0; s >>= 1) {
        if (g < s) red[threadIdx.x] = fmaxf(red[threadIdx.x], red[threadIdx.x + s * 32]);
        __syncthreads();
    }
    m = red[c];
    __syncthreads();
    float sum = 0.0f;
    for (int r = g; r < NU; r += 32) sum += __expf(k[r * KEYCH + c] - m);
    red[threadIdx.x] = sum;
    __syncthreads();
    for (int s = 16; s > 0; s >>= 1) {
        if (g < s) red[threadIdx.x] += red[threadIdx.x + s * 32];
        __syncthreads();
    }
    const float inv = 1.0f / red[c];
    for (int r = g; r < NU; r += 32) {
        int idx = r * KEYCH + c;
        k[idx] = __expf(k[idx] - m) * inv;
    }
}

// ---------------------------------------------------------------------------
// Phase 3: per-(item,head) softmax over HK=8 channels, in place on q[NI,32].
// ---------------------------------------------------------------------------
__global__ void gt_row_softmax_kernel(float* __restrict__ q) {
    int t = blockIdx.x * blockDim.x + threadIdx.x;   // NI*HEADS threads
    float* p = q + t * HK;
    float m = -INFINITY;
#pragma unroll
    for (int e = 0; e < HK; ++e) m = fmaxf(m, p[e]);
    float ex[HK], s = 0.0f;
#pragma unroll
    for (int e = 0; e < HK; ++e) { ex[e] = __expf(p[e] - m); s += ex[e]; }
    const float inv = 1.0f / s;
#pragma unroll
    for (int e = 0; e < HK; ++e) p[e] = ex[e] * inv;
}

// ---------------------------------------------------------------------------
// Phase 4: transpose-convert U[NU,32] fp32 -> Ut[32][NU] bf16.
// ---------------------------------------------------------------------------
__global__ void gt_pack_ut_kernel(const float* __restrict__ U,
                                  __bf16* __restrict__ Ut) {
    int t = blockIdx.x * blockDim.x + threadIdx.x;   // 32*NU threads
    int d = t >> 13, u = t & (NU - 1);
    Ut[t] = (__bf16)U[u * DMODEL + d];               // Ut[d*NU + u]
}

// ---------------------------------------------------------------------------
// One fused relu-attention step: 32 users x 32 items x 1 head.
//   8x V_WMMA_F32_16X16X4_F32  (scores, K=8 as two chained K=4)
//   4x V_WMMA_F32_16X16X32_BF16 (ctx^T @ U accumulate)
// f32 C-layout of S[u,i] == bf16 A-layout of S^T[i,u]: relu+cvt is lane-local.
// ---------------------------------------------------------------------------
union gt_bu { v16bf v; v8bf h2[2]; };

__device__ __forceinline__ void gt_mma_step(
    v2f a00, v2f a01, v2f a10, v2f a11,
    v2f bqA0, v2f bqA1, v2f bqB0, v2f bqB1,
    v16bf b0, v16bf b1,
    v8f& accA0, v8f& accA1, v8f& accB0, v8f& accB1) {
    v8f S0A = {}, S1A = {}, S0B = {}, S1B = {};
    S0A = __builtin_amdgcn_wmma_f32_16x16x4_f32(false, a00, false, bqA0, (short)0, S0A, false, false);
    S0A = __builtin_amdgcn_wmma_f32_16x16x4_f32(false, a01, false, bqA1, (short)0, S0A, false, false);
    S1A = __builtin_amdgcn_wmma_f32_16x16x4_f32(false, a10, false, bqA0, (short)0, S1A, false, false);
    S1A = __builtin_amdgcn_wmma_f32_16x16x4_f32(false, a11, false, bqA1, (short)0, S1A, false, false);
    S0B = __builtin_amdgcn_wmma_f32_16x16x4_f32(false, a00, false, bqB0, (short)0, S0B, false, false);
    S0B = __builtin_amdgcn_wmma_f32_16x16x4_f32(false, a01, false, bqB1, (short)0, S0B, false, false);
    S1B = __builtin_amdgcn_wmma_f32_16x16x4_f32(false, a10, false, bqB0, (short)0, S1B, false, false);
    S1B = __builtin_amdgcn_wmma_f32_16x16x4_f32(false, a11, false, bqB1, (short)0, S1B, false, false);

    v16bf abfA, abfB;
#pragma unroll
    for (int e = 0; e < 8; ++e) {
        abfA[e]     = (__bf16)fmaxf(S0A[e] - BIASC, 0.0f);
        abfA[8 + e] = (__bf16)fmaxf(S1A[e] - BIASC, 0.0f);
        abfB[e]     = (__bf16)fmaxf(S0B[e] - BIASC, 0.0f);
        abfB[8 + e] = (__bf16)fmaxf(S1B[e] - BIASC, 0.0f);
    }
    accA0 = __builtin_amdgcn_wmma_f32_16x16x32_bf16(false, abfA, false, b0, (short)0, accA0, false, false);
    accA1 = __builtin_amdgcn_wmma_f32_16x16x32_bf16(false, abfA, false, b1, (short)0, accA1, false, false);
    accB0 = __builtin_amdgcn_wmma_f32_16x16x32_bf16(false, abfB, false, b0, (short)0, accB0, false, false);
    accB1 = __builtin_amdgcn_wmma_f32_16x16x32_bf16(false, abfB, false, b1, (short)0, accB1, false, false);
}

// ---------------------------------------------------------------------------
// Phase 5: fused relu-attention main kernel.
// wave = one 32-item tile x one head x one u-half; 4 waves/block share the
// same head & u-range, so k and U chunks are staged once per block in LDS via
// GLOBAL_LOAD_ASYNC_TO_LDS_B128 (ASYNCcnt), double-buffered.
// ---------------------------------------------------------------------------
__global__ __launch_bounds__(128)
void gt_attn_kernel(const float* __restrict__ kbuf,
                    const float* __restrict__ qbuf,
                    const __bf16* __restrict__ Ut,
                    float* __restrict__ att) {          // [ZSPLIT][NI][128]
    const int lane  = threadIdx.x & 31;
    const int wave  = threadIdx.x >> 5;
    const int ibase = (blockIdx.x * 4 + wave) * ITILE;
    const int h     = blockIdx.y;
    const int uo    = blockIdx.z * (NU / ZSPLIT);
    const int lhalf = lane >> 4;
    const int lmod  = lane & 15;
    const int chOff = 8 * h + lhalf * 2;

    // Loop-invariant score-B operands (q rows, two 16-item subtiles).
    const float* qrA = qbuf + (ibase + lmod) * KEYCH + chOff;
    const float* qrB = qrA + 16 * KEYCH;
    const v2f bqA0 = *(const v2f*)qrA;
    const v2f bqA1 = *(const v2f*)(qrA + 4);
    const v2f bqB0 = *(const v2f*)qrB;
    const v2f bqB1 = *(const v2f*)(qrB + 4);

    v8f accA0 = {}, accA1 = {}, accB0 = {}, accB1 = {};

#if __has_builtin(__builtin_amdgcn_global_load_async_to_lds_b128)
    // -------- async-LDS staged path (gfx1250 ASYNCcnt) --------
    __shared__ __align__(16) __bf16 us[2][UTILE * DMODEL];  // 2 x 2 KB U chunk
    __shared__ __align__(16) float  ks[2][UTILE * HK];      // 2 x 1 KB k chunk

    // wave0, lane L stages: U row d=L (64B = 4xb128), k row uc+L head slice (32B = 2xb128)
    if (wave == 0) {
        const __bf16* gsrc = Ut + lane * NU + uo;
        const float*  ksrc = kbuf + (uo + lane) * KEYCH + 8 * h;
#pragma unroll
        for (int j = 0; j < 4; ++j)
            __builtin_amdgcn_global_load_async_to_lds_b128(
                (gt_gp)(gsrc + j * 8), (gt_lp)&us[0][lane * DMODEL + j * 8], 0, 0);
#pragma unroll
        for (int j = 0; j < 2; ++j)
            __builtin_amdgcn_global_load_async_to_lds_b128(
                (gt_gp)(ksrc + j * 4), (gt_lp)&ks[0][lane * HK + j * 4], 0, 0);
    }

    for (int c = 0; c < NCHUNK; ++c) {
        const int cur = c & 1;
        if (wave == 0) {
#if __has_builtin(__builtin_amdgcn_s_wait_asynccnt)
            __builtin_amdgcn_s_wait_asynccnt(0);
#else
            asm volatile("s_wait_asynccnt 0x0" ::: "memory");
#endif
        }
        __syncthreads();   // buf[cur] ready for all; prior reads of buf[cur^1] done
        if (wave == 0 && c + 1 < NCHUNK) {
            const int un = uo + (c + 1) * UTILE;
            const __bf16* gsrc = Ut + lane * NU + un;
            const float*  ksrc = kbuf + (un + lane) * KEYCH + 8 * h;
#pragma unroll
            for (int j = 0; j < 4; ++j)
                __builtin_amdgcn_global_load_async_to_lds_b128(
                    (gt_gp)(gsrc + j * 8), (gt_lp)&us[cur ^ 1][lane * DMODEL + j * 8], 0, 0);
#pragma unroll
            for (int j = 0; j < 2; ++j)
                __builtin_amdgcn_global_load_async_to_lds_b128(
                    (gt_gp)(ksrc + j * 4), (gt_lp)&ks[cur ^ 1][lane * HK + j * 4], 0, 0);
        }

        // score-A operands (k) from LDS
        const float* kp = &ks[cur][0];
        const v2f a00 = *(const v2f*)(kp + lmod * HK + lhalf * 2);
        const v2f a01 = *(const v2f*)(kp + lmod * HK + lhalf * 2 + 4);
        const v2f a10 = *(const v2f*)(kp + (lmod + 16) * HK + lhalf * 2);
        const v2f a11 = *(const v2f*)(kp + (lmod + 16) * HK + lhalf * 2 + 4);
        // bf16 B operands (U chunk) from LDS
        const __bf16* up = &us[cur][0];
        gt_bu b0, b1;
        b0.h2[0] = *(const v8bf*)(up + lmod * DMODEL + lhalf * 8);
        b0.h2[1] = *(const v8bf*)(up + lmod * DMODEL + lhalf * 8 + 16);
        b1.h2[0] = *(const v8bf*)(up + (lmod + 16) * DMODEL + lhalf * 8);
        b1.h2[1] = *(const v8bf*)(up + (lmod + 16) * DMODEL + lhalf * 8 + 16);

        gt_mma_step(a00, a01, a10, a11, bqA0, bqA1, bqB0, bqB1,
                    b0.v, b1.v, accA0, accA1, accB0, accB1);
    }
#else
    // -------- fallback: direct global loads + prefetch --------
    const __bf16* ub0 = Ut + (lmod)      * NU + lhalf * 8;
    const __bf16* ub1 = Ut + (lmod + 16) * NU + lhalf * 8;
    for (int c = 0; c < NCHUNK; ++c) {
        const int uc = uo + c * UTILE;
        const float* kr0 = kbuf + (uc + lmod) * KEYCH + chOff;
        const float* kr1 = kr0 + 16 * KEYCH;
        const v2f a00 = *(const v2f*)kr0;
        const v2f a01 = *(const v2f*)(kr0 + 4);
        const v2f a10 = *(const v2f*)kr1;
        const v2f a11 = *(const v2f*)(kr1 + 4);
        gt_bu b0, b1;
        b0.h2[0] = *(const v8bf*)(ub0 + uc);
        b0.h2[1] = *(const v8bf*)(ub0 + uc + 16);
        b1.h2[0] = *(const v8bf*)(ub1 + uc);
        b1.h2[1] = *(const v8bf*)(ub1 + uc + 16);
        __builtin_prefetch(ub0 + uc + UTILE, 0, 1);
        __builtin_prefetch(ub1 + uc + UTILE, 0, 1);
        gt_mma_step(a00, a01, a10, a11, bqA0, bqA1, bqB0, bqB1,
                    b0.v, b1.v, accA0, accA1, accB0, accB1);
    }
#endif

    // Store partial att[i, h*32 + d] for this u-half (head-major, stride 128).
    float* ao = att + (size_t)blockIdx.z * (NI * HEADS * DMODEL);
#pragma unroll
    for (int r = 0; r < 8; ++r) {
        const int rowA = ibase + r + lhalf * 8;
        ao[rowA * (HEADS * DMODEL) + h * DMODEL + lmod]             = accA0[r];
        ao[rowA * (HEADS * DMODEL) + h * DMODEL + 16 + lmod]        = accA1[r];
        ao[(rowA + 16) * (HEADS * DMODEL) + h * DMODEL + lmod]      = accB0[r];
        ao[(rowA + 16) * (HEADS * DMODEL) + h * DMODEL + 16 + lmod] = accB1[r];
    }
}

// ---------------------------------------------------------------------------
// Phase 6: sum the two u-halves + projection (128 -> 32) + bias + residual.
// ---------------------------------------------------------------------------
__global__ void gt_proj_kernel(const float* __restrict__ att,   // [2][NI][128]
                               const float* __restrict__ item,
                               const float* __restrict__ W,
                               const float* __restrict__ b,
                               float* __restrict__ out) {
    int t = blockIdx.x * blockDim.x + threadIdx.x;   // NI*32 threads
    int i = t >> 5, c = t & 31;
    const float* a0 = att + i * (HEADS * DMODEL);
    const float* a1 = a0 + NI * (HEADS * DMODEL);
    float acc = b[c];
#pragma unroll 4
    for (int j = 0; j < HEADS * DMODEL; ++j)
        acc = fmaf(a0[j] + a1[j], W[j * KEYCH + c], acc);
    out[t] = item[t] + acc;
}

// ---------------------------------------------------------------------------
// Launcher
// ---------------------------------------------------------------------------
extern "C" void kernel_launch(void* const* d_in, const int* in_sizes, int n_in,
                              void* d_out, int out_size, void* d_ws, size_t ws_size,
                              hipStream_t stream) {
    const float* U   = (const float*)d_in[0];   // [8192,32]
    const float* I   = (const float*)d_in[1];   // [8192,32]
    const float* kW  = (const float*)d_in[2];
    const float* kB  = (const float*)d_in[3];
    const float* qW  = (const float*)d_in[4];
    const float* qB  = (const float*)d_in[5];
    const float* rW  = (const float*)d_in[6];   // [128,32]
    const float* rB  = (const float*)d_in[7];
    float* out = (float*)d_out;

    char* ws = (char*)d_ws;
    float*  kbuf = (float*)(ws);                      // 1 MB   : keys -> k
    float*  qbuf = (float*)(ws + (1u << 20));         // 1 MB   : queries -> q
    float*  attb = (float*)(ws + (2u << 20));         // 8 MB   : att [2][8192][128]
    __bf16* ut   = (__bf16*)(ws + (10u << 20));       // 0.5 MB : U^T bf16 [32,8192]

    gt_dense_kernel<<<(NU * KEYCH) / 256, 256, 0, stream>>>(U, kW, kB, kbuf);
    gt_dense_kernel<<<(NI * KEYCH) / 256, 256, 0, stream>>>(I, qW, qB, qbuf);
    gt_col_softmax_kernel<<<1, 1024, 0, stream>>>(kbuf);
    gt_row_softmax_kernel<<<(NI * HEADS) / 256, 256, 0, stream>>>(qbuf);
    gt_pack_ut_kernel<<<(DMODEL * NU) / 256, 256, 0, stream>>>(U, ut);
    // 64 x 4 x 2 blocks, 4 waves each => 2048 waves, 12 WMMA / 32-user chunk
    gt_attn_kernel<<<dim3(NI / ITILE / 4, HEADS, ZSPLIT), 128, 0, stream>>>(kbuf, qbuf, ut, attb);
    gt_proj_kernel<<<(NI * KEYCH) / 256, 256, 0, stream>>>(attb, I, rW, rB, out);
}
